// MOE_17068200034824
// MI455X (gfx1250) — compile-verified, matrix operand order checked by reading, not run on hardware
//
#include <hip/hip_runtime.h>

// ---------------------------------------------------------------------------
// MoE forward for MI455X (gfx1250).
//   * noisy-top2 gating (wave32 butterfly reductions)
//   * one-time fp32->bf16 weight transpose (WMMA-friendly [N][K] layout)
//   * grouped GEMMs on v_wmma_f32_16x16x32_bf16; double-buffered LDS staged
//     with gfx1250 global_load_async_to_lds_b128 (ASYNCcnt). The pipeline is
//     unrolled 2x so buffer selection is compile-time -> fragment reads stay
//     ds_load_b128 (LDS pipe), not flat_load.
// Useful work: ~275 GFLOP vs ~0.8 GB traffic -> compute-bound on WMMA.
// ---------------------------------------------------------------------------

#define DEV_INLINE __device__ __forceinline__

typedef __bf16 bf16_t;
typedef __attribute__((ext_vector_type(16))) __bf16 v16bf;
typedef __attribute__((ext_vector_type(8)))  __bf16 v8bf;
typedef __attribute__((ext_vector_type(8)))  float  v8f;

constexpr int N_TOK = 8192;
constexpr int N_EXP = 8;
constexpr int DIM   = 1024;
constexpr int HID   = 4096;
constexpr int TOPK  = 2;
constexpr int ROWS  = N_TOK * TOPK;   // 16384 routed rows

// GEMM tiling: 128x128 tile, K-stage 64 (2 wmma K-steps), 256 thr = 8 waves.
constexpr int TM = 128;
constexpr int TN = 128;
constexpr int TK = 64;
constexpr int LPITCH = TK + 8;        // 72 elems = 144B row pitch, 16B-aligned chunks

struct alignas(8) bf16x4 { bf16_t v[4]; };

// --------------------------- CDNA5 async copy ------------------------------
// lds_byte: wave-relative LDS byte address (low 32 bits of the flat address,
// ISA 10.2).
DEV_INLINE void async_cp16(uint32_t lds_byte, const bf16_t* g) {
  asm volatile("global_load_async_to_lds_b128 %0, %1, off"
               :: "v"(lds_byte), "v"((unsigned long long)(uintptr_t)g)
               : "memory");
}
DEV_INLINE void wait_async0() { asm volatile("s_wait_asynccnt 0x0" ::: "memory"); }

// ---------------------------------------------------------------------------
// utility kernels
// ---------------------------------------------------------------------------
__global__ void zero_out_kernel(float4* __restrict__ out) {
  out[(size_t)blockIdx.x * blockDim.x + threadIdx.x] = make_float4(0.f, 0.f, 0.f, 0.f);
}

__global__ void zero_counts_kernel(int* __restrict__ counts) {
  if (threadIdx.x < N_EXP) counts[threadIdx.x] = 0;
}

// one-time weight convert+transpose: W [E][kdim][ndim] f32 -> Wt [E][ndim][kdim] bf16
constexpr int TPT = 64;               // transpose tile
constexpr int TPP = 66;               // LDS pitch (bank-conflict-free: gcd(33,64)=1)
__global__ __launch_bounds__(256)
void transpose_w_kernel(const float* __restrict__ W, bf16_t* __restrict__ Wt,
                        int kdim, int ndim) {
  __shared__ bf16_t Ls[TPT * TPP];
  const int e  = blockIdx.z;
  const int kb = blockIdx.x * TPT;
  const int nb = blockIdx.y * TPT;
  const float* src = W  + ((size_t)e * kdim + kb) * ndim + nb;
  bf16_t*      dst = Wt + ((size_t)e * ndim + nb) * kdim + kb;
  const int tid = threadIdx.x;
  {
    const int n  = tid & 63;
    const int k0 = tid >> 6;                       // 0..3
#pragma unroll
    for (int p = 0; p < 16; ++p) {
      const int k = k0 + p * 4;
      Ls[k * TPP + n] = (bf16_t)src[(size_t)k * ndim + n];   // coalesced in n
    }
  }
  __syncthreads();
  {
    const int k  = tid & 63;
    const int n0 = tid >> 6;
#pragma unroll
    for (int p = 0; p < 16; ++p) {
      const int n = n0 + p * 4;
      dst[(size_t)n * kdim + k] = Ls[k * TPP + n];           // coalesced in k
    }
  }
}

// ---------------------------------------------------------------------------
// gating: one wave (32 lanes) per token.
// ---------------------------------------------------------------------------
__global__ __launch_bounds__(256)
void gating_kernel(const float* __restrict__ x, const float* __restrict__ noise,
                   const float* __restrict__ Wg_w, const float* __restrict__ Wg_b,
                   const float* __restrict__ Wn_w, const float* __restrict__ Wn_b,
                   int* __restrict__ topk_idx, float* __restrict__ topk_gate,
                   int* __restrict__ counts) {
  const int lane = threadIdx.x & 31;
  const int wid  = threadIdx.x >> 5;
  const int t    = blockIdx.x * 8 + wid;

  float ag[N_EXP], an[N_EXP];
#pragma unroll
  for (int e = 0; e < N_EXP; ++e) { ag[e] = 0.f; an[e] = 0.f; }

  const float* xr = x + (size_t)t * DIM;
  for (int d = lane; d < DIM; d += 32) {
    const float xv = xr[d];
    const float4 g0 = *reinterpret_cast<const float4*>(Wg_w + (size_t)d * N_EXP);
    const float4 g1 = *reinterpret_cast<const float4*>(Wg_w + (size_t)d * N_EXP + 4);
    const float4 n0 = *reinterpret_cast<const float4*>(Wn_w + (size_t)d * N_EXP);
    const float4 n1 = *reinterpret_cast<const float4*>(Wn_w + (size_t)d * N_EXP + 4);
    ag[0] = fmaf(xv, g0.x, ag[0]); ag[1] = fmaf(xv, g0.y, ag[1]);
    ag[2] = fmaf(xv, g0.z, ag[2]); ag[3] = fmaf(xv, g0.w, ag[3]);
    ag[4] = fmaf(xv, g1.x, ag[4]); ag[5] = fmaf(xv, g1.y, ag[5]);
    ag[6] = fmaf(xv, g1.z, ag[6]); ag[7] = fmaf(xv, g1.w, ag[7]);
    an[0] = fmaf(xv, n0.x, an[0]); an[1] = fmaf(xv, n0.y, an[1]);
    an[2] = fmaf(xv, n0.z, an[2]); an[3] = fmaf(xv, n0.w, an[3]);
    an[4] = fmaf(xv, n1.x, an[4]); an[5] = fmaf(xv, n1.y, an[5]);
    an[6] = fmaf(xv, n1.z, an[6]); an[7] = fmaf(xv, n1.w, an[7]);
  }
#pragma unroll
  for (int m = 16; m >= 1; m >>= 1) {
#pragma unroll
    for (int e = 0; e < N_EXP; ++e) {
      ag[e] += __shfl_xor(ag[e], m, 32);
      an[e] += __shfl_xor(an[e], m, 32);
    }
  }
  if (lane == 0) {
    float h[N_EXP];
#pragma unroll
    for (int e = 0; e < N_EXP; ++e) {
      const float z  = an[e] + Wn_b[e];
      const float sp = (z > 20.f) ? z : log1pf(expf(z));     // softplus
      h[e] = ag[e] + Wg_b[e] + noise[(size_t)t * N_EXP + e] * sp;
    }
    int i0 = 0;
#pragma unroll
    for (int e = 1; e < N_EXP; ++e) if (h[e] > h[i0]) i0 = e;
    int i1 = (i0 == 0) ? 1 : 0;
#pragma unroll
    for (int e = 0; e < N_EXP; ++e) if (e != i0 && h[e] > h[i1]) i1 = e;
    const float p  = expf(h[i1] - h[i0]);
    const float g0 = 1.f / (1.f + p);
    const float g1 = p / (1.f + p);
    topk_idx[t * TOPK + 0]  = i0;
    topk_idx[t * TOPK + 1]  = i1;
    topk_gate[t * TOPK + 0] = g0;
    topk_gate[t * TOPK + 1] = g1;
    atomicAdd(&counts[i0], 1);
    atomicAdd(&counts[i1], 1);
  }
}

__global__ void scan_kernel(const int* __restrict__ counts,
                            int* __restrict__ offsets, int* __restrict__ cursors) {
  if (threadIdx.x == 0) {
    int s = 0;
#pragma unroll
    for (int e = 0; e < N_EXP; ++e) { offsets[e] = s; cursors[e] = s; s += counts[e]; }
    offsets[N_EXP] = s;
  }
}

__global__ void route_kernel(const int* __restrict__ topk_idx,
                             const float* __restrict__ topk_gate,
                             int* __restrict__ cursors,
                             int* __restrict__ row_token, float* __restrict__ row_gate) {
  const int i = blockIdx.x * blockDim.x + threadIdx.x;
  if (i >= ROWS) return;
  const int e   = topk_idx[i];
  const int pos = atomicAdd(&cursors[e], 1);
  row_token[pos] = i / TOPK;
  row_gate[pos]  = topk_gate[i];
}

__global__ __launch_bounds__(256)
void gather_kernel(const float* __restrict__ x, const int* __restrict__ row_token,
                   bf16_t* __restrict__ Xg) {
  const int r  = blockIdx.x;
  const int t  = row_token[r];
  const int d0 = threadIdx.x * 4;     // 256*4 == DIM
  const float4 v = *reinterpret_cast<const float4*>(x + (size_t)t * DIM + d0);
  bf16x4 o;
  o.v[0] = (bf16_t)v.x; o.v[1] = (bf16_t)v.y; o.v[2] = (bf16_t)v.z; o.v[3] = (bf16_t)v.w;
  *reinterpret_cast<bf16x4*>(Xg + (size_t)r * DIM + d0) = o;
}

// ---------------------------------------------------------------------------
// WMMA GEMM core. A: bf16 [valid x kdim] (routed rows). B: bf16 [TN x kdim]
// (pre-transposed weights, [n][k]). Both async-staged to LDS; fragments are
// contiguous 16B chunks of the 16-bit 16x32 WMMA layout (ISA 7.12.2).
// ---------------------------------------------------------------------------
DEV_INLINE v16bf ld_frag(const bf16_t* rowbase, int kh) {
  const v8bf lo = *reinterpret_cast<const v8bf*>(rowbase + kh * 8);
  const v8bf hi = *reinterpret_cast<const v8bf*>(rowbase + 16 + kh * 8);
  v16bf r;
#pragma unroll
  for (int i = 0; i < 8; ++i) { r[i] = lo[i]; r[i + 8] = hi[i]; }
  return r;
}

// issue one stage of async copies (A tile rows + B tile rows, 16B per lane x4)
DEV_INLINE void stage_async(uint32_t ldsA, uint32_t ldsB,
                            const bf16_t* __restrict__ Aglob,
                            const bf16_t* __restrict__ Bglob,
                            int kdim, int kb, int valid_rows, int r0, int c) {
#pragma unroll
  for (int p = 0; p < 4; ++p) {
    const int r = r0 + p * 32;
    const uint32_t lo = (uint32_t)(r * LPITCH + c) * 2u;
    if (r < valid_rows)
      async_cp16(ldsA + lo, Aglob + (size_t)r * kdim + kb + c);
    async_cp16(ldsB + lo, Bglob + (size_t)r * kdim + kb + c);
  }
}

// 2 wmma K-steps (16 WMMAs) from one LDS buffer pair
DEV_INLINE void compute_stage(const bf16_t* Ab, const bf16_t* Bb,
                              int wm, int wn, int lrow, int kh, v8f acc[4][2]) {
#pragma unroll
  for (int s = 0; s < 2; ++s) {
    v16bf af[4], bfr[2];
#pragma unroll
    for (int i = 0; i < 4; ++i)
      af[i] = ld_frag(Ab + (wm * 64 + i * 16 + lrow) * LPITCH + s * 32, kh);
#pragma unroll
    for (int j = 0; j < 2; ++j)
      bfr[j] = ld_frag(Bb + (wn * 32 + j * 16 + lrow) * LPITCH + s * 32, kh);
#pragma unroll
    for (int i = 0; i < 4; ++i)
#pragma unroll
      for (int j = 0; j < 2; ++j)
        acc[i][j] = __builtin_amdgcn_wmma_f32_16x16x32_bf16(
            false, af[i], false, bfr[j], (short)0, acc[i][j], false, false);
  }
}

DEV_INLINE void gemm_tile_async(const bf16_t* __restrict__ Aglob,
                                const bf16_t* __restrict__ Bglob, int kdim,
                                int valid_rows,
                                bf16_t* As0, bf16_t* As1,
                                bf16_t* Bs0, bf16_t* Bs1, v8f acc[4][2]) {
  const int tid  = threadIdx.x;
  const int lane = tid & 31;
  const int wid  = tid >> 5;
  const int wm   = wid & 1;
  const int wn   = wid >> 1;
  const int lrow = lane & 15;
  const int kh   = lane >> 4;
  const int r0   = tid >> 3;          // staging row (0..31, x4 passes)
  const int c    = (tid & 7) * 8;     // staging 16B chunk (elements)

  const uint32_t ldsA0 = (uint32_t)(uintptr_t)As0;
  const uint32_t ldsA1 = (uint32_t)(uintptr_t)As1;
  const uint32_t ldsB0 = (uint32_t)(uintptr_t)Bs0;
  const uint32_t ldsB1 = (uint32_t)(uintptr_t)Bs1;

  // zero LDS once for padded A rows (async copies never write them)
  if (valid_rows < TM) {
    const uint4 z = make_uint4(0u, 0u, 0u, 0u);
    const int nchunks = (TM - valid_rows) * 8;
    for (int zi = tid; zi < nchunks; zi += 256) {
      const int r  = valid_rows + (zi >> 3);
      const int cc = (zi & 7) * 8;
      *reinterpret_cast<uint4*>(As0 + r * LPITCH + cc) = z;
      *reinterpret_cast<uint4*>(As1 + r * LPITCH + cc) = z;
    }
  }

  // prologue: stage k-block 0 into buffer 0
  stage_async(ldsA0, ldsB0, Aglob, Bglob, kdim, 0, valid_rows, r0, c);
  wait_async0();
  __syncthreads();

  // pipeline unrolled 2x: buffer selection is compile-time, frag reads stay
  // on the LDS pipe (ds_load_b128). kdim/TK is even (16 or 64).
  for (int kb = 0; kb < kdim; kb += 2 * TK) {
    stage_async(ldsA1, ldsB1, Aglob, Bglob, kdim, kb + TK, valid_rows, r0, c);
    compute_stage(As0, Bs0, wm, wn, lrow, kh, acc);
    wait_async0();
    __syncthreads();

    if (kb + 2 * TK < kdim)
      stage_async(ldsA0, ldsB0, Aglob, Bglob, kdim, kb + 2 * TK, valid_rows, r0, c);
    compute_stage(As1, Bs1, wm, wn, lrow, kh, acc);
    wait_async0();
    __syncthreads();
  }
}

// fc1: hid = relu(Xg @ w1[e] + b1[e]) -> bf16 workspace
__global__ __launch_bounds__(256)
void fc1_kernel(const bf16_t* __restrict__ Xg,
                const bf16_t* __restrict__ Wt1, const float* __restrict__ b1,
                const int* __restrict__ offsets, bf16_t* __restrict__ Hidb) {
  __shared__ __align__(16) bf16_t As0[TM * LPITCH], As1[TM * LPITCH];
  __shared__ __align__(16) bf16_t Bs0[TN * LPITCH], Bs1[TN * LPITCH];

  const int e       = blockIdx.z;
  const int rbeg    = offsets[e];
  const int rend    = offsets[e + 1];
  const int rowbase = rbeg + blockIdx.x * TM;
  if (rowbase >= rend) return;
  const int valid = min(TM, rend - rowbase);
  const int nbase = blockIdx.y * TN;

  v8f acc[4][2];
#pragma unroll
  for (int i = 0; i < 4; ++i)
#pragma unroll
    for (int j = 0; j < 2; ++j)
#pragma unroll
      for (int q = 0; q < 8; ++q) acc[i][j][q] = 0.f;

  gemm_tile_async(Xg + (size_t)rowbase * DIM,
                  Wt1 + ((size_t)e * HID + nbase) * DIM, DIM, valid,
                  As0, As1, Bs0, Bs1, acc);

  const int lane = threadIdx.x & 31;
  const int wid  = threadIdx.x >> 5;
  const int wm = wid & 1, wn = wid >> 1;
  const int mlo = (lane >> 4) * 8;     // C layout: VGPR r -> M = r + (l>>4)*8
  const int nn  = lane & 15;
#pragma unroll
  for (int i = 0; i < 4; ++i)
#pragma unroll
    for (int j = 0; j < 2; ++j) {
      const int   ncol = nbase + wn * 32 + j * 16 + nn;
      const float bias = b1[(size_t)e * HID + ncol];
#pragma unroll
      for (int r = 0; r < 8; ++r) {
        const int m = wm * 64 + i * 16 + mlo + r;
        if (m < valid) {
          float v = acc[i][j][r] + bias;
          v = fmaxf(v, 0.f);
          Hidb[(size_t)(rowbase + m) * HID + ncol] = (bf16_t)v;
        }
      }
    }
}

// fc2: out[token] += gate * (hid @ w2[e] + b2[e])
__global__ __launch_bounds__(256)
void fc2_kernel(const bf16_t* __restrict__ Hidb,
                const bf16_t* __restrict__ Wt2, const float* __restrict__ b2,
                const int* __restrict__ offsets,
                const int* __restrict__ row_token, const float* __restrict__ row_gate,
                float* __restrict__ out) {
  __shared__ __align__(16) bf16_t As0[TM * LPITCH], As1[TM * LPITCH];
  __shared__ __align__(16) bf16_t Bs0[TN * LPITCH], Bs1[TN * LPITCH];

  const int e       = blockIdx.z;
  const int rbeg    = offsets[e];
  const int rend    = offsets[e + 1];
  const int rowbase = rbeg + blockIdx.x * TM;
  if (rowbase >= rend) return;
  const int valid = min(TM, rend - rowbase);
  const int nbase = blockIdx.y * TN;

  v8f acc[4][2];
#pragma unroll
  for (int i = 0; i < 4; ++i)
#pragma unroll
    for (int j = 0; j < 2; ++j)
#pragma unroll
      for (int q = 0; q < 8; ++q) acc[i][j][q] = 0.f;

  gemm_tile_async(Hidb + (size_t)rowbase * HID,
                  Wt2 + ((size_t)e * DIM + nbase) * HID, HID, valid,
                  As0, As1, Bs0, Bs1, acc);

  const int lane = threadIdx.x & 31;
  const int wid  = threadIdx.x >> 5;
  const int wm = wid & 1, wn = wid >> 1;
  const int mlo = (lane >> 4) * 8;
  const int nn  = lane & 15;
#pragma unroll
  for (int i = 0; i < 4; ++i)
#pragma unroll
    for (int j = 0; j < 2; ++j) {
      const int   ncol = nbase + wn * 32 + j * 16 + nn;
      const float bias = b2[(size_t)e * DIM + ncol];
#pragma unroll
      for (int r = 0; r < 8; ++r) {
        const int m = wm * 64 + i * 16 + mlo + r;
        if (m < valid) {
          const int   row = rowbase + m;
          const float v   = (acc[i][j][r] + bias) * row_gate[row];
          atomicAdd(out + (size_t)row_token[row] * DIM + ncol, v);
        }
      }
    }
}

// ---------------------------------------------------------------------------
// host launch
// ---------------------------------------------------------------------------
extern "C" void kernel_launch(void* const* d_in, const int* in_sizes, int n_in,
                              void* d_out, int out_size, void* d_ws, size_t ws_size,
                              hipStream_t stream) {
  const float* x     = (const float*)d_in[0];
  const float* noise = (const float*)d_in[1];
  const float* Wg_w  = (const float*)d_in[2];
  const float* Wg_b  = (const float*)d_in[3];
  const float* Wn_w  = (const float*)d_in[4];
  const float* Wn_b  = (const float*)d_in[5];
  const float* w1    = (const float*)d_in[6];
  const float* b1    = (const float*)d_in[7];
  const float* w2    = (const float*)d_in[8];
  const float* b2    = (const float*)d_in[9];
  float* out = (float*)d_out;

  char* base = (char*)d_ws;
  size_t off = 0;
  auto carve = [&](size_t bytes) -> char* {
    char* p = base + off;
    off = (off + bytes + 255) & ~(size_t)255;
    return p;
  };
  int*    counts    = (int*)   carve(N_EXP * sizeof(int));
  int*    offsets   = (int*)   carve((N_EXP + 1) * sizeof(int));
  int*    cursors   = (int*)   carve(N_EXP * sizeof(int));
  int*    topk_idx  = (int*)   carve((size_t)ROWS * sizeof(int));
  float*  topk_gate = (float*) carve((size_t)ROWS * sizeof(float));
  int*    row_token = (int*)   carve((size_t)ROWS * sizeof(int));
  float*  row_gate  = (float*) carve((size_t)ROWS * sizeof(float));
  bf16_t* Xg        = (bf16_t*)carve((size_t)ROWS * DIM * sizeof(bf16_t));
  bf16_t* Hidb      = (bf16_t*)carve((size_t)ROWS * HID * sizeof(bf16_t));
  bf16_t* Wt1       = (bf16_t*)carve((size_t)N_EXP * DIM * HID * sizeof(bf16_t));
  bf16_t* Wt2       = (bf16_t*)carve((size_t)N_EXP * HID * DIM * sizeof(bf16_t));
  if (off > ws_size) return;

  // 1. zero output + expert counts
  zero_out_kernel<<<(N_TOK * DIM / 4) / 256, 256, 0, stream>>>((float4*)out);
  zero_counts_kernel<<<1, 32, 0, stream>>>(counts);

  // 2. weight convert+transpose to WMMA-friendly bf16 [E][N][K]
  transpose_w_kernel<<<dim3(DIM / TPT, HID / TPT, N_EXP), 256, 0, stream>>>(
      w1, Wt1, DIM, HID);
  transpose_w_kernel<<<dim3(HID / TPT, DIM / TPT, N_EXP), 256, 0, stream>>>(
      w2, Wt2, HID, DIM);

  // 3. gating (one wave per token)
  gating_kernel<<<N_TOK / 8, 256, 0, stream>>>(x, noise, Wg_w, Wg_b, Wn_w, Wn_b,
                                               topk_idx, topk_gate, counts);

  // 4. offsets scan (E=8, trivial)
  scan_kernel<<<1, 32, 0, stream>>>(counts, offsets, cursors);

  // 5. route tokens into expert-grouped rows
  route_kernel<<<(ROWS + 255) / 256, 256, 0, stream>>>(topk_idx, topk_gate, cursors,
                                                       row_token, row_gate);

  // 6. gather routed rows to bf16
  gather_kernel<<<ROWS, 256, 0, stream>>>(x, row_token, Xg);

  // 7. fc1 grouped GEMM: [rows x D] @ [D x H] + relu
  fc1_kernel<<<dim3(N_TOK / TM, HID / TN, N_EXP), 256, 0, stream>>>(Xg, Wt1, b1,
                                                                    offsets, Hidb);

  // 8. fc2 grouped GEMM + gate-weighted scatter-add
  fc2_kernel<<<dim3(N_TOK / TM, DIM / TN, N_EXP), 256, 0, stream>>>(Hidb, Wt2, b2,
                                                                    offsets, row_token,
                                                                    row_gate, out);
}
